// GINEncoder_41515153883619
// MI455X (gfx1250) — matrix operand order repaired and checked by reading, not compiled, not run on hardware
//
#include <hip/hip_runtime.h>
#include <hip/hip_bf16.h>

// ---------------------------------------------------------------------------
// GIN encoder: 3 x [ scatter-add aggregate -> MLP(128x128 GEMM, ReLU, GEMM)
//                    -> BatchNorm(train stats) -> ReLU (except last) ]
// GEMMs use v_wmma_f32_16x16x32_bf16 (wave32 CDNA5 WMMA).
// ---------------------------------------------------------------------------

#define GIN_N_NODES 50000
#define GIN_N_EDGES 800000
#define GIN_DIM     128
#define GIN_LAYERS  3
#define GIN_BN_EPS  1e-5f
#define GIN_TILES   (GIN_N_NODES / 16)   // 3125 exact

typedef __bf16 v16bf __attribute__((ext_vector_type(16)));
typedef float  v8f   __attribute__((ext_vector_type(8)));

// ---------------------------------------------------------------------------
// One-time: convert the 6 weight matrices (f32, row-major [K=128][N=128]) into
// bf16 B-fragment layout. Fragment f = kt*8+nt (kt: K-chunk of 32, nt: N-tile
// of 16). Per ISA 7.12.2 (16-bit B 32x16): lane l holds column n=l&15,
// K-values k = 16*(j>>3) + 8*(l>>4) + (j&7) for j=0..15, so each lane's fragment is 16
// contiguous bf16 (32B) in this layout -> one aligned vector load in the GEMM.
// ---------------------------------------------------------------------------
__global__ void gin_prep_weights(const float* __restrict__ W1s,
                                 const float* __restrict__ W2s,
                                 __bf16* __restrict__ wfrag) {
    int tid = blockIdx.x * blockDim.x + threadIdx.x;
    if (tid >= 2 * GIN_LAYERS * GIN_DIM * GIN_DIM) return;
    int mi = tid / (GIN_DIM * GIN_DIM);     // matrix index 0..5
    int r  = tid % (GIN_DIM * GIN_DIM);
    int f  = r >> 9;                        // fragment 0..31
    int l  = (r >> 4) & 31;                 // lane
    int j  = r & 15;                        // element within lane fragment
    int kt = f >> 3, nt = f & 7;
    int n  = nt * 16 + (l & 15);
    int k  = kt * 32 + ((j >> 3) * 16) + ((l >> 4) * 8) + (j & 7);
    int layer = mi >> 1;
    const float* W = (mi & 1) ? (W2s + (size_t)layer * GIN_DIM * GIN_DIM)
                              : (W1s + (size_t)layer * GIN_DIM * GIN_DIM);
    wfrag[tid] = (__bf16)W[k * GIN_DIM + n];
}

// agg = xin  (h = x + sum gets the "x +" for free), and zero BN stats buffer.
__global__ void gin_copy_zero(const float4* __restrict__ xin,
                              float4* __restrict__ agg,
                              float* __restrict__ stats) {
    int tid = blockIdx.x * blockDim.x + threadIdx.x;
    if (tid < 2 * GIN_DIM) stats[tid] = 0.0f;
    if (tid < GIN_N_NODES * GIN_DIM / 4) agg[tid] = xin[tid];
}

// 32 lanes per edge; lane c handles features 4c..4c+3 (float4 gather + 4
// global f32 atomics into agg[dst]).
__global__ void gin_scatter_edges(const float* __restrict__ xin,
                                  const long long* __restrict__ src,
                                  const long long* __restrict__ dst,
                                  float* __restrict__ agg) {
    long long tid = (long long)blockIdx.x * blockDim.x + threadIdx.x;
    int lane = (int)(tid & 31);
    long long e = tid >> 5;
    if (e >= GIN_N_EDGES) return;
    int s = (int)src[e];
    int d = (int)dst[e];
    const float4 v = *(const float4*)(xin + (long long)s * GIN_DIM + lane * 4);
    float* o = agg + (long long)d * GIN_DIM + lane * 4;
    atomicAdd(o + 0, v.x);
    atomicAdd(o + 1, v.y);
    atomicAdd(o + 2, v.z);
    atomicAdd(o + 3, v.w);
}

// A-fragment (16-bit A 16x32, ISA 7.12.2): lane l -> row m = l&15 of the tile;
// element j -> k = kt*32 + 16*(j>=8) + 8*(l>>4) + (j&7). Two contiguous 8-wide
// runs per lane per kt.
// C/D (f32 16x16): element r -> m = r + 8*(l>>4), n = l&15.

// GEMM1: h1 = relu(agg @ W1 + b1), f32 in, bf16 out (feeds GEMM2's A).
__global__ void __launch_bounds__(128)
gin_gemm_mlp1(const float* __restrict__ A, const __bf16* __restrict__ wfrag,
              const float* __restrict__ bias, __bf16* __restrict__ out) {
    int wave = (blockIdx.x * blockDim.x + threadIdx.x) >> 5;
    int l = threadIdx.x & 31;
    if (wave >= GIN_TILES) return;
    int row0 = wave * 16;
    int mlane = l & 15, hi = l >> 4;
    const float* arow = A + (long long)(row0 + mlane) * GIN_DIM + hi * 8;

    v8f acc[8];
#pragma unroll
    for (int nt = 0; nt < 8; ++nt) acc[nt] = (v8f){0, 0, 0, 0, 0, 0, 0, 0};

#pragma unroll
    for (int kt = 0; kt < 4; ++kt) {
        v16bf afrag;
        const float* ap = arow + kt * 32;
#pragma unroll
        for (int j = 0; j < 8; ++j) afrag[j] = (__bf16)ap[j];
#pragma unroll
        for (int j = 0; j < 8; ++j) afrag[8 + j] = (__bf16)ap[16 + j];
        const __bf16* wf = wfrag + ((size_t)(kt * 8) * 32 + l) * 16;
#pragma unroll
        for (int nt = 0; nt < 8; ++nt) {
            v16bf bfrag = *(const v16bf*)(wf + (size_t)nt * 32 * 16);
            acc[nt] = __builtin_amdgcn_wmma_f32_16x16x32_bf16(
                false, afrag, false, bfrag, (short)0, acc[nt], false, false);
        }
    }

#pragma unroll
    for (int nt = 0; nt < 8; ++nt) {
        int n = nt * 16 + mlane;
        float b = bias[n];
#pragma unroll
        for (int r = 0; r < 8; ++r) {
            float v = acc[nt][r] + b;
            v = v > 0.0f ? v : 0.0f;
            out[(long long)(row0 + r + 8 * hi) * GIN_DIM + n] = (__bf16)v;
        }
    }
}

// GEMM2: h2 = h1 @ W2 + b2 (f32 out), plus per-column sum / sum-of-squares
// reduced across the two lane halves (shfl_xor 16) and atomically added to
// stats[0:128] / stats[128:256].
__global__ void __launch_bounds__(128)
gin_gemm_mlp2(const __bf16* __restrict__ A, const __bf16* __restrict__ wfrag,
              const float* __restrict__ bias, float* __restrict__ out,
              float* __restrict__ stats) {
    int wave = (blockIdx.x * blockDim.x + threadIdx.x) >> 5;
    int l = threadIdx.x & 31;
    if (wave >= GIN_TILES) return;
    int row0 = wave * 16;
    int mlane = l & 15, hi = l >> 4;
    const __bf16* arow = A + (long long)(row0 + mlane) * GIN_DIM + hi * 8;

    v8f acc[8];
#pragma unroll
    for (int nt = 0; nt < 8; ++nt) acc[nt] = (v8f){0, 0, 0, 0, 0, 0, 0, 0};

#pragma unroll
    for (int kt = 0; kt < 4; ++kt) {
        v16bf afrag;
        const __bf16* ap = arow + kt * 32;
#pragma unroll
        for (int j = 0; j < 8; ++j) afrag[j] = ap[j];
#pragma unroll
        for (int j = 0; j < 8; ++j) afrag[8 + j] = ap[16 + j];
        const __bf16* wf = wfrag + ((size_t)(kt * 8) * 32 + l) * 16;
#pragma unroll
        for (int nt = 0; nt < 8; ++nt) {
            v16bf bfrag = *(const v16bf*)(wf + (size_t)nt * 32 * 16);
            acc[nt] = __builtin_amdgcn_wmma_f32_16x16x32_bf16(
                false, afrag, false, bfrag, (short)0, acc[nt], false, false);
        }
    }

#pragma unroll
    for (int nt = 0; nt < 8; ++nt) {
        int n = nt * 16 + mlane;
        float b = bias[n];
        float s1 = 0.0f, s2 = 0.0f;
#pragma unroll
        for (int r = 0; r < 8; ++r) {
            float v = acc[nt][r] + b;
            out[(long long)(row0 + r + 8 * hi) * GIN_DIM + n] = v;
            s1 += v;
            s2 += v * v;
        }
        s1 += __shfl_xor(s1, 16, 32);   // combine the two half-lanes (rows 0-7 / 8-15)
        s2 += __shfl_xor(s2, 16, 32);
        if (hi == 0) {
            atomicAdd(stats + n, s1);
            atomicAdd(stats + GIN_DIM + n, s2);
        }
    }
}

// BatchNorm (biased variance: E[h^2]-E[h]^2) + affine + optional ReLU.
__global__ void gin_bn_apply(const float* __restrict__ h,
                             const float* __restrict__ stats,
                             const float* __restrict__ gamma,
                             const float* __restrict__ beta,
                             float* __restrict__ out, int do_relu) {
    __shared__ float s_scale[GIN_DIM];
    __shared__ float s_shift[GIN_DIM];
    int t = threadIdx.x;
    if (t < GIN_DIM) {
        const float invn = 1.0f / (float)GIN_N_NODES;
        float mean = stats[t] * invn;
        float var  = stats[GIN_DIM + t] * invn - mean * mean;
        float inv  = rsqrtf(var + GIN_BN_EPS);
        float sc   = gamma[t] * inv;
        s_scale[t] = sc;
        s_shift[t] = beta[t] - mean * sc;
    }
    __syncthreads();
    const long long total = (long long)GIN_N_NODES * GIN_DIM;
    for (long long i = (long long)blockIdx.x * blockDim.x + t; i < total;
         i += (long long)gridDim.x * blockDim.x) {
        int c = (int)(i & (GIN_DIM - 1));
        float v = h[i] * s_scale[c] + s_shift[c];
        if (do_relu) v = v > 0.0f ? v : 0.0f;
        out[i] = v;
    }
}

extern "C" void kernel_launch(void* const* d_in, const int* in_sizes, int n_in,
                              void* d_out, int out_size, void* d_ws, size_t ws_size,
                              hipStream_t stream) {
    (void)in_sizes; (void)n_in; (void)out_size; (void)ws_size;

    const float*     x      = (const float*)d_in[0];
    const long long* edge   = (const long long*)d_in[1];   // int64 [2, E]
    const float*     W1s    = (const float*)d_in[2];
    const float*     b1s    = (const float*)d_in[3];
    const float*     W2s    = (const float*)d_in[4];
    const float*     b2s    = (const float*)d_in[5];
    const float*     gammas = (const float*)d_in[6];
    const float*     betas  = (const float*)d_in[7];
    float*           outp   = (float*)d_out;

    // Workspace carve-out (256B aligned).
    char* ws = (char*)d_ws;
    size_t off = 0;
    auto carve = [&](size_t bytes) -> void* {
        void* p = ws + off;
        off += (bytes + 255) & ~(size_t)255;
        return p;
    };
    const size_t nd = (size_t)GIN_N_NODES * GIN_DIM;
    float*  agg   = (float*)carve(nd * sizeof(float));
    float*  h2    = (float*)carve(nd * sizeof(float));
    float*  bufA  = (float*)carve(nd * sizeof(float));
    __bf16* h1bf  = (__bf16*)carve(nd * sizeof(__bf16));
    __bf16* wfrag = (__bf16*)carve((size_t)2 * GIN_LAYERS * GIN_DIM * GIN_DIM * sizeof(__bf16));
    float*  stats = (float*)carve(2 * GIN_DIM * sizeof(float));

    // One-time weight conversion into WMMA B-fragment layout.
    {
        int nthr = 2 * GIN_LAYERS * GIN_DIM * GIN_DIM;   // 98304
        gin_prep_weights<<<(nthr + 255) / 256, 256, 0, stream>>>(W1s, W2s, wfrag);
    }

    const long long* src = edge;
    const long long* dst = edge + GIN_N_EDGES;

    for (int layer = 0; layer < GIN_LAYERS; ++layer) {
        const float* xin   = (layer == 0) ? x : bufA;
        float*       bnout = (layer == GIN_LAYERS - 1) ? outp : bufA;

        gin_copy_zero<<<(GIN_N_NODES * GIN_DIM / 4 + 255) / 256, 256, 0, stream>>>(
            (const float4*)xin, (float4*)agg, stats);

        long long sthreads = (long long)GIN_N_EDGES * 32;
        gin_scatter_edges<<<(unsigned)((sthreads + 255) / 256), 256, 0, stream>>>(
            xin, src, dst, agg);

        gin_gemm_mlp1<<<(GIN_TILES * 32 + 127) / 128, 128, 0, stream>>>(
            agg, wfrag + (size_t)(layer * 2) * GIN_DIM * GIN_DIM,
            b1s + layer * GIN_DIM, h1bf);

        gin_gemm_mlp2<<<(GIN_TILES * 32 + 127) / 128, 128, 0, stream>>>(
            h1bf, wfrag + (size_t)(layer * 2 + 1) * GIN_DIM * GIN_DIM,
            b2s + layer * GIN_DIM, h2, stats);

        gin_bn_apply<<<2048, 256, 0, stream>>>(
            h2, stats, gammas + layer * GIN_DIM, betas + layer * GIN_DIM,
            bnout, layer < GIN_LAYERS - 1 ? 1 : 0);
    }
}
